// SpatioTemporalTransformerRedux_8126078124253
// MI455X (gfx1250) — matrix operand order repaired and checked by reading, not compiled, not run on hardware
//
#include <hip/hip_runtime.h>
#include <hip/hip_bf16.h>

typedef __attribute__((ext_vector_type(16))) __bf16 v16bf;
typedef __attribute__((ext_vector_type(8)))  float  v8f;

#define D_MODEL 256
#define NHEAD   8
#define DH      32
#define NLAYER  4
#define DFF     1024
#define PATCH   8
#define CCH     2
#define DATA_D  128
#define PNUM    16
#define BB      4
#define TT      64
#define LSEQ    1024
#define QKVS    768
#define NEGV    -1000000.0f

// ---------------------------------------------------------------------------
// fragment loaders (wave32 WMMA; contiguous k order chosen by this file)
// A frag: lane holds row (lane&15); elem j -> K = j + 16*(lane>>4)
// B frag: lane holds col (lane&15); elem j -> K = j + 16*(lane>>4)
// ---------------------------------------------------------------------------
__device__ inline v16bf load_A_frag(const __bf16* A, int stride, int m_base, int kk, int lane) {
    const int hf = lane >> 4;
    const int m  = m_base + (lane & 15);
    const __bf16* p = A + (size_t)m * stride + kk + 16 * hf;
    v16bf a;
#pragma unroll
    for (int j = 0; j < 8; ++j) a[j] = p[j];
#pragma unroll
    for (int j = 0; j < 8; ++j) a[8 + j] = p[8 + j];
    return a;
}

__device__ inline v16bf load_B_frag(const __bf16* W, int stride, int n_base, int kk, int lane) {
    const int hf = lane >> 4;
    const int n  = n_base + (lane & 15);
    const __bf16* p = W + (size_t)n * stride + kk + 16 * hf;
    v16bf b;
#pragma unroll
    for (int j = 0; j < 16; ++j) b[j] = p[j];
    return b;
}

// ---------------------------------------------------------------------------
// generic WMMA GEMM: out[M,N] = A[M,K](bf16) * W[N,K]^T(bf16) + bias, opt relu
// each wave computes a 64x16 output block (4 M-tiles) -> B fragment reused 4x.
// M must be a multiple of 64, N of 16, K of 32.
// ---------------------------------------------------------------------------
__global__ __launch_bounds__(256)
void gemm_bt(const __bf16* __restrict__ A, const __bf16* __restrict__ W,
             const float* __restrict__ bias, float* outF, __bf16* outB,
             int M, int N, int K, int relu) {
    const int lane = threadIdx.x & 31;
    const int wave = threadIdx.x >> 5;
    const int tiles_n = N >> 4;
    const int total   = (M >> 6) * tiles_n;
    const int tile_id = blockIdx.x * 8 + wave;
    if (tile_id >= total) return;
    const int tm = (tile_id / tiles_n) << 6;
    const int tn = (tile_id % tiles_n) << 4;

    v8f acc0 = {}, acc1 = {}, acc2 = {}, acc3 = {};
    for (int kk = 0; kk < K; kk += 32) {
        v16bf b = load_B_frag(W, K, tn, kk, lane);
        // prefetch next K-step (speculative; dropped if OOB)
        __builtin_prefetch(W + (size_t)(tn + (lane & 15)) * K + kk + 32, 0, 1);
        __builtin_prefetch(A + (size_t)(tm + (lane & 15)) * K + kk + 32, 0, 1);
        v16bf a0 = load_A_frag(A, K, tm,      kk, lane);
        v16bf a1 = load_A_frag(A, K, tm + 16, kk, lane);
        v16bf a2 = load_A_frag(A, K, tm + 32, kk, lane);
        v16bf a3 = load_A_frag(A, K, tm + 48, kk, lane);
        acc0 = __builtin_amdgcn_wmma_f32_16x16x32_bf16(false, a0, false, b, (short)0, acc0, false, false);
        acc1 = __builtin_amdgcn_wmma_f32_16x16x32_bf16(false, a1, false, b, (short)0, acc1, false, false);
        acc2 = __builtin_amdgcn_wmma_f32_16x16x32_bf16(false, a2, false, b, (short)0, acc2, false, false);
        acc3 = __builtin_amdgcn_wmma_f32_16x16x32_bf16(false, a3, false, b, (short)0, acc3, false, false);
    }
    const int hf = lane >> 4;
    const int n  = tn + (lane & 15);
    const float bv = bias ? bias[n] : 0.0f;
    v8f accs[4] = {acc0, acc1, acc2, acc3};
#pragma unroll
    for (int t = 0; t < 4; ++t) {
#pragma unroll
        for (int r = 0; r < 8; ++r) {
            const int m = tm + 16 * t + r + 8 * hf;
            float v = accs[t][r] + bv;
            if (relu) v = v > 0.0f ? v : 0.0f;
            const size_t idx = (size_t)m * N + n;
            if (outF) outF[idx] = v;
            if (outB) outB[idx] = (__bf16)v;
        }
    }
}

// ---------------------------------------------------------------------------
// flash attention: one wave handles one (b, h, 16-query tile).
// scores stay in accumulators; P transposed D->A layout through LDS;
// V tile staged in LDS via coalesced row loads (no strided global gathers).
// ---------------------------------------------------------------------------
__global__ __launch_bounds__(32)
void attn_kernel(const __bf16* __restrict__ qkv, const int* __restrict__ rowmask,
                 const int* __restrict__ padk, __bf16* __restrict__ out) {
    const int lane = threadIdx.x;
    const int hf   = lane >> 4;
    const int ln   = lane & 15;
    int blk = blockIdx.x;
    const int QT = LSEQ / 16;
    const int qt = blk % QT; blk /= QT;
    const int h  = blk % NHEAD;
    const int b  = blk / NHEAD;

    const size_t base = (size_t)b * LSEQ * QKVS;
    const __bf16* Q  = qkv + base +       h * DH;
    const __bf16* Kp = qkv + base + 256 + h * DH;
    const __bf16* Vp = qkv + base + 512 + h * DH;
    const int q0 = qt * 16;

    // Q fragment (A layout), K-dim = dh = 32 (single wmma depth)
    v16bf qa;
    {
        const __bf16* p = Q + (size_t)(q0 + ln) * QKVS + 16 * hf;
#pragma unroll
        for (int j = 0; j < 8; ++j) qa[j] = p[j];
#pragma unroll
        for (int j = 0; j < 8; ++j) qa[8 + j] = p[8 + j];
    }
    int rm[8];
#pragma unroll
    for (int r = 0; r < 8; ++r) rm[r] = rowmask[b * LSEQ + q0 + r + 8 * hf];

    float mrow[8], lrow[8];
    v8f o0 = {}, o1 = {};
#pragma unroll
    for (int r = 0; r < 8; ++r) { mrow[r] = -3.0e38f; lrow[r] = 0.0f; }

    __shared__ __bf16 Pt[16][34];
    __shared__ __bf16 Vl[32][36];
    const float scale = 0.17677669529663687f; // 1/sqrt(32)

    for (int kt = 0; kt < LSEQ; kt += 32) {
        // stage V tile: lane loads key row (kt+lane), 32 bf16 = 64B coalesced
        {
            const __bf16* vrow = Vp + (size_t)(kt + lane) * QKVS;
#pragma unroll
            for (int j = 0; j < 32; ++j) Vl[lane][j] = vrow[j];
        }
        v16bf kb0, kb1;
        {
            const __bf16* p0 = Kp + (size_t)(kt + ln) * QKVS + 16 * hf;
            const __bf16* p1 = Kp + (size_t)(kt + 16 + ln) * QKVS + 16 * hf;
#pragma unroll
            for (int j = 0; j < 16; ++j) kb0[j] = p0[j];
#pragma unroll
            for (int j = 0; j < 16; ++j) kb1[j] = p1[j];
        }
        v8f zz = {};
        v8f s0 = __builtin_amdgcn_wmma_f32_16x16x32_bf16(false, qa, false, kb0, (short)0, zz, false, false);
        v8f s1 = __builtin_amdgcn_wmma_f32_16x16x32_bf16(false, qa, false, kb1, (short)0, zz, false, false);

        const int c0 = kt + ln, c1 = kt + 16 + ln;
        const int pk0 = padk[b * LSEQ + c0];
        const int pk1 = padk[b * LSEQ + c1];
        float p0v[8], p1v[8];
#pragma unroll
        for (int r = 0; r < 8; ++r) {
            const int q = q0 + r + 8 * hf;
            float v0 = s0[r] * scale + ((rm[r] && q != c0) ? NEGV : 0.0f) + (pk0 ? NEGV : 0.0f);
            float v1 = s1[r] * scale + ((rm[r] && q != c1) ? NEGV : 0.0f) + (pk1 ? NEGV : 0.0f);
            float tmax = fmaxf(v0, v1);
#pragma unroll
            for (int off = 1; off < 16; off <<= 1) tmax = fmaxf(tmax, __shfl_xor(tmax, off, 16));
            const float mnew  = fmaxf(mrow[r], tmax);
            const float alpha = __expf(mrow[r] - mnew);
            const float e0 = __expf(v0 - mnew);
            const float e1 = __expf(v1 - mnew);
            float rs = e0 + e1;
#pragma unroll
            for (int off = 1; off < 16; off <<= 1) rs += __shfl_xor(rs, off, 16);
            lrow[r] = lrow[r] * alpha + rs;
            mrow[r] = mnew;
            o0[r] *= alpha; o1[r] *= alpha;
            p0v[r] = e0; p1v[r] = e1;
        }
        __syncthreads();
#pragma unroll
        for (int r = 0; r < 8; ++r) {
            Pt[r + 8 * hf][ln]      = (__bf16)p0v[r];
            Pt[r + 8 * hf][ln + 16] = (__bf16)p1v[r];
        }
        __syncthreads();
        v16bf pa;
#pragma unroll
        for (int j = 0; j < 8; ++j) pa[j]     = Pt[ln][16 * hf + j];
#pragma unroll
        for (int j = 0; j < 8; ++j) pa[8 + j] = Pt[ln][16 * hf + 8 + j];

        // V fragments (B layout) gathered from LDS: elem j -> key = 16*hf + j
        v16bf vb0, vb1;
#pragma unroll
        for (int j = 0; j < 16; ++j) vb0[j] = Vl[16 * hf + j][ln];
#pragma unroll
        for (int j = 0; j < 16; ++j) vb1[j] = Vl[16 * hf + j][ln + 16];

        o0 = __builtin_amdgcn_wmma_f32_16x16x32_bf16(false, pa, false, vb0, (short)0, o0, false, false);
        o1 = __builtin_amdgcn_wmma_f32_16x16x32_bf16(false, pa, false, vb1, (short)0, o1, false, false);
        __syncthreads();
    }
#pragma unroll
    for (int r = 0; r < 8; ++r) {
        const int m = q0 + r + 8 * hf;
        const float inv = 1.0f / lrow[r];
        out[((size_t)b * LSEQ + m) * D_MODEL + h * DH + ln]      = (__bf16)(o0[r] * inv);
        out[((size_t)b * LSEQ + m) * D_MODEL + h * DH + 16 + ln] = (__bf16)(o1[r] * inv);
    }
}

// ---------------------------------------------------------------------------
// elementwise / small kernels
// ---------------------------------------------------------------------------
__global__ void f2bf_kernel(const float* __restrict__ s, __bf16* __restrict__ d, int n) {
    int i = blockIdx.x * 256 + threadIdx.x;
    if (i < n) d[i] = (__bf16)s[i];
}

__global__ __launch_bounds__(256)
void frame_pad_kernel(const float* __restrict__ img, int* __restrict__ fp) {
    __shared__ int sh[256];
    const int bt = blockIdx.x, t = threadIdx.x;
    const float* p = img + (size_t)bt * 2048;
    int all = 1;
    for (int i = t; i < 2048; i += 256) all &= (p[i] == -9999.0f) ? 1 : 0;
    sh[t] = all; __syncthreads();
    for (int s = 128; s > 0; s >>= 1) { if (t < s) sh[t] &= sh[t + s]; __syncthreads(); }
    if (t == 0) fp[bt] = sh[0];
}

__global__ void masks_kernel(const float* __restrict__ img, const int* __restrict__ fp,
                             int* __restrict__ rowmask, int* __restrict__ padk) {
    const int idx = blockIdx.x * 256 + threadIdx.x;
    if (idx >= BB * LSEQ) return;
    const int b = idx >> 10, l = idx & 1023, t = l >> 4, p = l & 15;
    const int ph = p >> 2, pw = p & 3;
    const float* base = img + ((size_t)b * TT + t) * 2048;
    int allnan = 1;
    for (int c = 0; c < CCH && allnan; ++c)
        for (int py = 0; py < 8 && allnan; ++py)
            for (int px = 0; px < 8; ++px) {
                float v = base[c * 1024 + (ph * 8 + py) * 32 + pw * 8 + px];
                if (!(v != v)) { allnan = 0; break; }
            }
    const int f = fp[b * TT + t];
    padk[idx]    = f;
    rowmask[idx] = (allnan | f);
}

__global__ void patchify_kernel(const float* __restrict__ img, const float* __restrict__ tok,
                                __bf16* __restrict__ xp) {
    const size_t idx = (size_t)blockIdx.x * 256 + threadIdx.x;
    if (idx >= (size_t)BB * LSEQ * DATA_D) return;
    const int d = idx & 127; const size_t row = idx >> 7;
    const int b = row >> 10, l = row & 1023, t = l >> 4, p = l & 15;
    const int ph = p >> 2, pw = p & 3;
    const int c = d >> 6, py = (d >> 3) & 7, px = d & 7;
    float v = img[((size_t)(b * TT + t) * CCH + c) * 1024 + (ph * 8 + py) * 32 + pw * 8 + px];
    if (v != v) { float tk = tok[c]; v = fminf(fmaxf(tk, -10.f), 10.f); }
    v = fminf(fmaxf(v, -10.f), 10.f);
    xp[idx] = (__bf16)v;
}

__global__ void time_mlp(const float* __restrict__ acq,
                         const float* __restrict__ w1, const float* __restrict__ b1,
                         const float* __restrict__ w2, const float* __restrict__ b2,
                         const float* __restrict__ w3, const float* __restrict__ b3,
                         float* __restrict__ temb) {
    const int idx = blockIdx.x * 64 + threadIdx.x;
    if (idx >= BB * TT) return;
    float tin = acq[idx];
    if (tin != tin) tin = 0.0f;
    float h1[64];
#pragma unroll
    for (int i = 0; i < 64; ++i) { float v = tin * w1[i] + b1[i]; h1[i] = v > 0 ? v : 0; }
    float h2[128];
    for (int i = 0; i < 128; ++i) {
        float a = b2[i];
        for (int j = 0; j < 64; ++j) a += w2[i * 64 + j] * h1[j];
        h2[i] = a > 0 ? a : 0;
    }
    for (int d = 0; d < D_MODEL; ++d) {
        float a = b3[d];
        for (int j = 0; j < 128; ++j) a += w3[d * 128 + j] * h2[j];
        temb[(size_t)idx * D_MODEL + d] = a;
    }
}

__global__ void embed_epilogue(float* zF, __bf16* zB,
                               const float* __restrict__ spat, const float* __restrict__ temb,
                               const float* __restrict__ pademb, const int* __restrict__ padk) {
    const size_t idx = (size_t)blockIdx.x * 256 + threadIdx.x;
    const int d = idx & 255; const size_t row = idx >> 8;
    const int b = row >> 10, l = row & 1023, t = l >> 4, p = l & 15;
    float v;
    if (padk[row]) v = pademb[d];
    else v = zF[idx] + spat[p * D_MODEL + d] + temb[(size_t)(b * TT + t) * D_MODEL + d];
    zF[idx] = v; zB[idx] = (__bf16)v;
}

__global__ __launch_bounds__(256)
void add_ln(const float* x, const float* r, const float* __restrict__ w,
            const float* __restrict__ bsa, float* outF, __bf16* outB) {
    __shared__ float red[256];
    const int row = blockIdx.x, t = threadIdx.x;
    const size_t idx = (size_t)row * D_MODEL + t;
    const float v = x[idx] + r[idx];
    red[t] = v; __syncthreads();
    for (int s = 128; s > 0; s >>= 1) { if (t < s) red[t] += red[t + s]; __syncthreads(); }
    const float mean = red[0] * (1.0f / 256.0f);
    __syncthreads();
    const float d = v - mean;
    red[t] = d * d; __syncthreads();
    for (int s = 128; s > 0; s >>= 1) { if (t < s) red[t] += red[t + s]; __syncthreads(); }
    const float var = red[0] * (1.0f / 256.0f);
    const float o = d * rsqrtf(var + 1e-5f) * w[t] + bsa[t];
    outF[idx] = o; outB[idx] = (__bf16)o;
}

__global__ void gather_last(const __bf16* __restrict__ zB, __bf16* __restrict__ zl) {
    const int idx = blockIdx.x * 256 + threadIdx.x;
    const int d = idx & 255; const int row = idx >> 8;   // row = b*16 + p
    const int b = row >> 4, p = row & 15;
    zl[idx] = zB[((size_t)b * LSEQ + (TT - 1) * PNUM + p) * D_MODEL + d];
}

__global__ void final_out(const float* __restrict__ meanF, const float* __restrict__ lvF,
                          float* __restrict__ out) {
    const int idx = blockIdx.x * 256 + threadIdx.x;
    if (idx >= BB * CCH * 32 * 32) return;
    const int w = idx & 31, hh = (idx >> 5) & 31, c = (idx >> 10) & 1, b = idx >> 11;
    const int ph = hh >> 3, py = hh & 7, pw = w >> 3, px = w & 7;
    const int row = b * PNUM + ph * 4 + pw;
    const int d = c * 64 + py * 8 + px;
    float m  = meanF[row * DATA_D + d];
    float lv = lvF[row * DATA_D + d];
    out[idx]        = fminf(fmaxf(m, -10.f), 10.f);
    out[8192 + idx] = fminf(fmaxf(lv, -10.f), 5.f);
}

// ---------------------------------------------------------------------------
extern "C" void kernel_launch(void* const* d_in, const int* in_sizes, int n_in,
                              void* d_out, int out_size, void* d_ws, size_t ws_size,
                              hipStream_t stream) {
    const float* img     = (const float*)d_in[0];
    const float* acq     = (const float*)d_in[1];
    const float* ntok    = (const float*)d_in[2];
    const float* pademb  = (const float*)d_in[3];
    const float* emb_w   = (const float*)d_in[4];
    const float* emb_b   = (const float*)d_in[5];
    const float* spat    = (const float*)d_in[6];
    const float* tw1     = (const float*)d_in[7];
    const float* tb1     = (const float*)d_in[8];
    const float* tw2     = (const float*)d_in[9];
    const float* tb2     = (const float*)d_in[10];
    const float* tw3     = (const float*)d_in[11];
    const float* tb3     = (const float*)d_in[12];
    const float* inp_w   = (const float*)d_in[13];
    const float* inp_b   = (const float*)d_in[14];
    const float* outp_w  = (const float*)d_in[15];
    const float* outp_b  = (const float*)d_in[16];
    const float* l1_w    = (const float*)d_in[17];
    const float* l1_b    = (const float*)d_in[18];
    const float* l2_w    = (const float*)d_in[19];
    const float* l2_b    = (const float*)d_in[20];
    const float* n1_w    = (const float*)d_in[21];
    const float* n1_b    = (const float*)d_in[22];
    const float* n2_w    = (const float*)d_in[23];
    const float* n2_b    = (const float*)d_in[24];
    const float* m_w1    = (const float*)d_in[25];
    const float* m_b1    = (const float*)d_in[26];
    const float* m_w2    = (const float*)d_in[27];
    const float* m_b2    = (const float*)d_in[28];
    const float* v_w1    = (const float*)d_in[29];
    const float* v_b1    = (const float*)d_in[30];
    const float* v_w2    = (const float*)d_in[31];
    const float* v_b2    = (const float*)d_in[32];

    size_t off = 0;
    auto alloc = [&](size_t bytes) -> void* {
        void* p = (char*)d_ws + off;
        off += (bytes + 255) & ~(size_t)255;
        return p;
    };
    __bf16* w_emb  = (__bf16*)alloc((size_t)D_MODEL * DATA_D * 2);
    __bf16* w_inp  = (__bf16*)alloc((size_t)NLAYER * QKVS * D_MODEL * 2);
    __bf16* w_outp = (__bf16*)alloc((size_t)NLAYER * D_MODEL * D_MODEL * 2);
    __bf16* w_l1   = (__bf16*)alloc((size_t)NLAYER * DFF * D_MODEL * 2);
    __bf16* w_l2   = (__bf16*)alloc((size_t)NLAYER * D_MODEL * DFF * 2);
    __bf16* w_m1   = (__bf16*)alloc((size_t)DFF * D_MODEL * 2);
    __bf16* w_m2   = (__bf16*)alloc((size_t)DATA_D * DFF * 2);
    __bf16* w_v1   = (__bf16*)alloc((size_t)DFF * D_MODEL * 2);
    __bf16* w_v2   = (__bf16*)alloc((size_t)DATA_D * DFF * 2);
    __bf16* xpB    = (__bf16*)alloc((size_t)BB * LSEQ * DATA_D * 2);
    int*    fpad   = (int*)alloc((size_t)BB * TT * 4);
    int*    rowm   = (int*)alloc((size_t)BB * LSEQ * 4);
    int*    padk   = (int*)alloc((size_t)BB * LSEQ * 4);
    float*  temb   = (float*)alloc((size_t)BB * TT * D_MODEL * 4);
    float*  zF     = (float*)alloc((size_t)BB * LSEQ * D_MODEL * 4);
    __bf16* zB     = (__bf16*)alloc((size_t)BB * LSEQ * D_MODEL * 2);
    __bf16* qkvB   = (__bf16*)alloc((size_t)BB * LSEQ * QKVS * 2);
    __bf16* attnB  = (__bf16*)alloc((size_t)BB * LSEQ * D_MODEL * 2);
    float*  tmpF   = (float*)alloc((size_t)BB * LSEQ * D_MODEL * 4);
    __bf16* hB     = (__bf16*)alloc((size_t)BB * LSEQ * DFF * 2);
    __bf16* zlB    = (__bf16*)alloc((size_t)BB * PNUM * D_MODEL * 2);
    float*  meanF  = (float*)alloc((size_t)BB * PNUM * DATA_D * 4);
    float*  lvF    = (float*)alloc((size_t)BB * PNUM * DATA_D * 4);

    auto cvt = [&](const float* s, __bf16* d, int n) {
        f2bf_kernel<<<(n + 255) / 256, 256, 0, stream>>>(s, d, n);
    };
    cvt(emb_w,  w_emb,  D_MODEL * DATA_D);
    cvt(inp_w,  w_inp,  NLAYER * QKVS * D_MODEL);
    cvt(outp_w, w_outp, NLAYER * D_MODEL * D_MODEL);
    cvt(l1_w,   w_l1,   NLAYER * DFF * D_MODEL);
    cvt(l2_w,   w_l2,   NLAYER * D_MODEL * DFF);
    cvt(m_w1,   w_m1,   DFF * D_MODEL);
    cvt(m_w2,   w_m2,   DATA_D * DFF);
    cvt(v_w1,   w_v1,   DFF * D_MODEL);
    cvt(v_w2,   w_v2,   DATA_D * DFF);

    frame_pad_kernel<<<BB * TT, 256, 0, stream>>>(img, fpad);
    masks_kernel<<<(BB * LSEQ + 255) / 256, 256, 0, stream>>>(img, fpad, rowm, padk);
    patchify_kernel<<<(BB * LSEQ * DATA_D + 255) / 256, 256, 0, stream>>>(img, ntok, xpB);
    time_mlp<<<(BB * TT + 63) / 64, 64, 0, stream>>>(acq, tw1, tb1, tw2, tb2, tw3, tb3, temb);

    auto gemm = [&](const __bf16* A, const __bf16* W, const float* bias,
                    float* oF, __bf16* oB, int M, int N, int K, int relu) {
        const int tiles = (M / 64) * (N / 16);
        gemm_bt<<<(tiles + 7) / 8, 256, 0, stream>>>(A, W, bias, oF, oB, M, N, K, relu);
    };

    const int Mrows = BB * LSEQ; // 4096
    gemm(xpB, w_emb, emb_b, zF, nullptr, Mrows, D_MODEL, DATA_D, 0);
    embed_epilogue<<<Mrows, 256, 0, stream>>>(zF, zB, spat, temb, pademb, padk);

    for (int l = 0; l < NLAYER; ++l) {
        gemm(zB, w_inp + (size_t)l * QKVS * D_MODEL, inp_b + l * QKVS,
             nullptr, qkvB, Mrows, QKVS, D_MODEL, 0);
        attn_kernel<<<BB * NHEAD * (LSEQ / 16), 32, 0, stream>>>(qkvB, rowm, padk, attnB);
        gemm(attnB, w_outp + (size_t)l * D_MODEL * D_MODEL, outp_b + l * D_MODEL,
             tmpF, nullptr, Mrows, D_MODEL, D_MODEL, 0);
        add_ln<<<Mrows, 256, 0, stream>>>(zF, tmpF, n1_w + l * D_MODEL, n1_b + l * D_MODEL, zF, zB);
        gemm(zB, w_l1 + (size_t)l * DFF * D_MODEL, l1_b + l * DFF,
             nullptr, hB, Mrows, DFF, D_MODEL, 1);
        gemm(hB, w_l2 + (size_t)l * D_MODEL * DFF, l2_b + l * D_MODEL,
             tmpF, nullptr, Mrows, D_MODEL, DFF, 0);
        add_ln<<<Mrows, 256, 0, stream>>>(zF, tmpF, n2_w + l * D_MODEL, n2_b + l * D_MODEL, zF, zB);
    }

    // heads: only last frame's 16 patches per batch -> 64 rows
    gather_last<<<(BB * PNUM * D_MODEL) / 256, 256, 0, stream>>>(zB, zlB);
    const int Mh = BB * PNUM; // 64
    gemm(zlB, w_m1, m_b1, nullptr, hB, Mh, DFF, D_MODEL, 1);
    gemm(hB,  w_m2, m_b2, meanF, nullptr, Mh, DATA_D, DFF, 0);
    gemm(zlB, w_v1, v_b1, nullptr, hB, Mh, DFF, D_MODEL, 1);
    gemm(hB,  w_v2, v_b2, lvF, nullptr, Mh, DATA_D, DFF, 0);

    final_out<<<(BB * CCH * 32 * 32 + 255) / 256, 256, 0, stream>>>(meanF, lvF, (float*)d_out);
}
